// QKVProjector_69638599737735
// MI455X (gfx1250) — matrix operand order repaired and checked by reading, not compiled, not run on hardware
//
#include <hip/hip_runtime.h>

typedef __bf16  bf16;
typedef __bf16  v4bf  __attribute__((ext_vector_type(4)));
typedef __bf16  v8bf  __attribute__((ext_vector_type(8)));
typedef __bf16  v16bf __attribute__((ext_vector_type(16)));
typedef float   v8f   __attribute__((ext_vector_type(8)));

#define BT_TOTAL 2048   // B*T
#define NHEADS   16     // N
#define INDIM    1024   // IN
#define D3DIM    192    // D3
#define KEXP     4      // K
#define BM       128    // rows per block tile
#define KC       32     // K-chunk (one WMMA K)
#define XS       48     // LDS row stride in bf16 (96B, 16B aligned)
#define WS       48
#define NCG      12     // 192 / 16 column groups

__device__ __forceinline__ v16bf combine16(v8bf a, v8bf b) {
    union { v8bf h[2]; v16bf v; } u;
    u.h[0] = a; u.h[1] = b;
    return u.v;
}

__device__ __forceinline__ void split_bf16(float f, bf16& hi, bf16& lo) {
    hi = (bf16)f;
    lo = (bf16)(f - (float)hi);
}

__global__ __launch_bounds__(256)
void qkv_wmma_bf16x3_kernel(const float* __restrict__ x,
                            const float* __restrict__ w,
                            const long long* __restrict__ perm,
                            const long long* __restrict__ invperm,
                            const int* __restrict__ phase_p,
                            float* __restrict__ out)
{
    __shared__ bf16 Xh[BM * XS];
    __shared__ bf16 Xl[BM * XS];
    __shared__ bf16 Wth[D3DIM * WS];   // transposed: [o][k]
    __shared__ bf16 Wtl[D3DIM * WS];

    const int tid  = threadIdx.x;
    const int lane = tid & 31;
    const int wv   = tid >> 5;      // wave 0..7
    const int half = lane >> 4;     // 0/1
    const int l16  = lane & 15;

    // which (k, n) GEMM this block handles
    const int kn = blockIdx.y;
    const int k  = kn >> 4;
    const int n  = kn & 15;
    const int ph = phase_p[0];
    const int h  = (int)perm[(size_t)(ph * KEXP + k) * NHEADS + n];
    const int g  = (int)invperm[(size_t)(ph * KEXP + k) * NHEADS + h];

    const int row0 = blockIdx.x * BM;   // global bt row of this tile

    const float* __restrict__ xg = x + (size_t)g * INDIM;                       // + bt*N*IN + i
    const float* __restrict__ wg = w + (size_t)(k * NHEADS + h) * INDIM * D3DIM; // + i*D3 + o

    v8f acc[NCG];
#pragma unroll
    for (int cg = 0; cg < NCG; ++cg) acc[cg] = (v8f){0.f,0.f,0.f,0.f,0.f,0.f,0.f,0.f};

    float4 xr[4];
    float4 wr[6];

    // ---- prefetch K-tile 0 into registers ----
#pragma unroll
    for (int j = 0; j < 4; ++j) {
        int idx = j * 256 + tid;
        int r = idx >> 3, c4 = idx & 7;                 // 8 float4 per row of 32
        xr[j] = *reinterpret_cast<const float4*>(
            xg + (size_t)(row0 + r) * (NHEADS * INDIM) + c4 * 4);
    }
#pragma unroll
    for (int j = 0; j < 6; ++j) {
        int idx = j * 256 + tid;
        int r = idx / 48, c4 = idx % 48;                // 48 float4 per row of 192
        wr[j] = *reinterpret_cast<const float4*>(
            wg + (size_t)r * D3DIM + c4 * 4);
    }

    const int NT = INDIM / KC;   // 32 K-tiles
    for (int kt = 0; kt < NT; ++kt) {
        __syncthreads();   // previous tile's consumers done

        // ---- stage registers -> LDS with f32 -> (bf16 hi, bf16 lo) split ----
#pragma unroll
        for (int j = 0; j < 4; ++j) {
            int idx = j * 256 + tid;
            int r = idx >> 3, c4 = idx & 7;
            float4 v = xr[j];
            bf16 h0,h1,h2,h3,l0,l1,l2,l3;
            split_bf16(v.x, h0, l0); split_bf16(v.y, h1, l1);
            split_bf16(v.z, h2, l2); split_bf16(v.w, h3, l3);
            v4bf hv = {h0,h1,h2,h3};
            v4bf lv = {l0,l1,l2,l3};
            *reinterpret_cast<v4bf*>(&Xh[r * XS + c4 * 4]) = hv;
            *reinterpret_cast<v4bf*>(&Xl[r * XS + c4 * 4]) = lv;
        }
#pragma unroll
        for (int j = 0; j < 6; ++j) {
            int idx = j * 256 + tid;
            int r = idx / 48, c4 = idx % 48;
            float4 v = wr[j];
            float c[4] = {v.x, v.y, v.z, v.w};
#pragma unroll
            for (int e = 0; e < 4; ++e) {
                bf16 hh, ll;
                split_bf16(c[e], hh, ll);
                int o = c4 * 4 + e;
                Wth[o * WS + r] = hh;   // transposed store
                Wtl[o * WS + r] = ll;
            }
        }

        __syncthreads();   // tile ready in LDS

        // ---- prefetch next K-tile into registers (overlaps with WMMA) ----
        if (kt + 1 < NT) {
            const int koff = (kt + 1) * KC;
#pragma unroll
            for (int j = 0; j < 4; ++j) {
                int idx = j * 256 + tid;
                int r = idx >> 3, c4 = idx & 7;
                xr[j] = *reinterpret_cast<const float4*>(
                    xg + (size_t)(row0 + r) * (NHEADS * INDIM) + koff + c4 * 4);
            }
#pragma unroll
            for (int j = 0; j < 6; ++j) {
                int idx = j * 256 + tid;
                int r = idx / 48, c4 = idx % 48;
                wr[j] = *reinterpret_cast<const float4*>(
                    wg + (size_t)(koff + r) * D3DIM + c4 * 4);
            }
        }

        // ---- compute: 3-product bf16 split GEMM on this K=32 chunk ----
        // A fragment (16x32 bf16): lanes 0-15 K=0..7|16..23, lanes 16-31 K=8..15|24..31
        const bf16* xph = &Xh[(wv * 16 + l16) * XS];
        const bf16* xpl = &Xl[(wv * 16 + l16) * XS];
        v16bf ah = combine16(*reinterpret_cast<const v8bf*>(xph + 8 * half),
                             *reinterpret_cast<const v8bf*>(xph + 16 + 8 * half));
        v16bf al = combine16(*reinterpret_cast<const v8bf*>(xpl + 8 * half),
                             *reinterpret_cast<const v8bf*>(xpl + 16 + 8 * half));
#pragma unroll
        for (int cg = 0; cg < NCG; ++cg) {
            // B fragment (32x16 bf16): lane = column, lanes 0-15 K=0..7|16..23, 16-31 K=8..15|24..31
            const bf16* wph = &Wth[(cg * 16 + l16) * WS + 8 * half];
            const bf16* wpl = &Wtl[(cg * 16 + l16) * WS + 8 * half];
            v16bf bh = combine16(*reinterpret_cast<const v8bf*>(wph),
                                 *reinterpret_cast<const v8bf*>(wph + 16));
            v16bf bl = combine16(*reinterpret_cast<const v8bf*>(wpl),
                                 *reinterpret_cast<const v8bf*>(wpl + 16));
            acc[cg] = __builtin_amdgcn_wmma_f32_16x16x32_bf16(
                false, ah, false, bh, (short)0, acc[cg], false, false);
            acc[cg] = __builtin_amdgcn_wmma_f32_16x16x32_bf16(
                false, ah, false, bl, (short)0, acc[cg], false, false);
            acc[cg] = __builtin_amdgcn_wmma_f32_16x16x32_bf16(
                false, al, false, bh, (short)0, acc[cg], false, false);
        }
    }

    // ---- epilogue: C/D layout -> out[(bt*K*N + kn)*192 + o] ----
    // VGPR v: lanes 0-15 -> M=v, lanes 16-31 -> M=v+8; N = l16
    const size_t outRowStride = (size_t)KEXP * NHEADS * D3DIM;  // 12288
#pragma unroll
    for (int cg = 0; cg < NCG; ++cg) {
        int col = cg * 16 + l16;
#pragma unroll
        for (int v = 0; v < 8; ++v) {
            int m = row0 + wv * 16 + v + 8 * half;
            out[(size_t)m * outRowStride + (size_t)kn * D3DIM + col] = acc[cg][v];
        }
    }
}

extern "C" void kernel_launch(void* const* d_in, const int* in_sizes, int n_in,
                              void* d_out, int out_size, void* d_ws, size_t ws_size,
                              hipStream_t stream) {
    (void)in_sizes; (void)n_in; (void)out_size; (void)d_ws; (void)ws_size;
    const float*      x       = (const float*)d_in[0];
    const float*      w       = (const float*)d_in[1];
    const long long*  perm    = (const long long*)d_in[2];
    const long long*  invperm = (const long long*)d_in[3];
    const int*        phase   = (const int*)d_in[4];
    float*            out     = (float*)d_out;

    dim3 grid(BT_TOTAL / BM, KEXP * NHEADS);   // 16 x 64
    qkv_wmma_bf16x3_kernel<<<grid, dim3(256), 0, stream>>>(
        x, w, perm, invperm, phase, out);
}
